// CrossAttention_lucas_17334488007436
// MI455X (gfx1250) — matrix operand order repaired and checked by reading, not compiled, not run on hardware
//
#include <hip/hip_runtime.h>
#include <hip/hip_bf16.h>

typedef _Float16 f16;
typedef __attribute__((ext_vector_type(16))) _Float16 v16h;
typedef __attribute__((ext_vector_type(8)))  _Float16 v8h;
typedef __attribute__((ext_vector_type(8)))  float    v8f;
typedef __attribute__((ext_vector_type(4)))  unsigned u32x4;

#define BATCH 16
#define CDIM  256
#define HEADS 4
#define DHEAD 64
#define NTOK  1024
#define SCALE 10.0f

// CDNA5 async global->LDS DMA (tracked by ASYNCcnt).  Set to 0 to fall back
// to the register round-trip copy.
#define USE_ASYNC 1

// ---------------------------------------------------------------------------
// 16-byte global -> LDS copy.  Async path: GLOBAL_LOAD_ASYNC_TO_LDS_B128
// (ISA 15.18.3 op 98): VDST = LDS byte address VGPR, VADDR = 64-bit address.
// ---------------------------------------------------------------------------
__device__ __forceinline__ void cp16(void* l, const void* g) {
#if USE_ASYNC
    unsigned lo = (unsigned)(size_t)(__attribute__((address_space(3))) void*)l;
    unsigned long long ga = (unsigned long long)g;
    asm volatile("global_load_async_to_lds_b128 %0, %1, off"
                 :: "v"(lo), "v"(ga) : "memory");
#else
    *(u32x4*)l = *(const u32x4*)g;
#endif
}
__device__ __forceinline__ void cp16_wait() {
#if USE_ASYNC
    asm volatile("s_wait_asynccnt 0x0" ::: "memory");
#endif
}

// ---------------------------------------------------------------------------
// Fragment helpers.  Per CDNA5 ISA 7.12.2 (wave32):
//  A 16x32 f16: lane l (m=l%16, g=l/16) holds A[m][8g..8g+7] ++ A[m][16+8g..+7]
//  B 32x16 f16: lane l (n=l%16, g=l/16) holds B[8g..8g+7][n] ++ B[16+8g..+7][n]
// Row-major A-tiles and K-major B-tiles -> every fragment is 2 x ds_load_b128.
// ---------------------------------------------------------------------------
__device__ __forceinline__ v16h cat8(v8h lo, v8h hi) {
    return __builtin_shufflevector(lo, hi, 0, 1, 2, 3, 4, 5, 6, 7,
                                   8, 9, 10, 11, 12, 13, 14, 15);
}
__device__ __forceinline__ v16h afrag32(const _Float16* A, int m, int g) {
    return cat8(*(const v8h*)(A + m * 32 + 8 * g),
                *(const v8h*)(A + m * 32 + 16 + 8 * g));
}
__device__ __forceinline__ v16h bfragT(const _Float16* Bt, int rs, int n,
                                       int g, int k0) {
    return cat8(*(const v8h*)(Bt + n * rs + k0 + 8 * g),
                *(const v8h*)(Bt + n * rs + k0 + 16 + 8 * g));
}
__device__ __forceinline__ v8f wmma_f16(v16h a, v16h b, v8f c) {
    return __builtin_amdgcn_wmma_f32_16x16x32_f16(false, a, false, b,
                                                  (short)0, c, false, false);
}

// ---------------------------------------------------------------------------
// Stage 1a: f32 -> f16 (weights, row-major: A operand)
// ---------------------------------------------------------------------------
__global__ void k_f32_to_f16(const float* __restrict__ src,
                             _Float16* __restrict__ dst, int n) {
    int i = blockIdx.x * blockDim.x + threadIdx.x;
    if (i < n) dst[i] = (_Float16)src[i];
}

// ---------------------------------------------------------------------------
// Stage 1b: transpose-convert: f32 [b][256][1024] -> f16 [b][1024][256]
// (K-major layout for the B operand of every GEMM)
// ---------------------------------------------------------------------------
__global__ __launch_bounds__(256) void k_transpose_f16(
    const float* __restrict__ src, _Float16* __restrict__ dst) {
    __shared__ _Float16 tile[32][33];
    const int n0 = blockIdx.x * 32, c0 = blockIdx.y * 32, bb = blockIdx.z;
    const int tx = threadIdx.x, ty = threadIdx.y;  // 32 x 8
    const float* s = src + (size_t)bb * CDIM * NTOK;
    _Float16* d = dst + (size_t)bb * NTOK * CDIM;
#pragma unroll
    for (int u = 0; u < 4; ++u)
        tile[ty + 8 * u][tx] = (_Float16)s[(c0 + ty + 8 * u) * NTOK + n0 + tx];
    __syncthreads();
#pragma unroll
    for (int u = 0; u < 4; ++u)
        d[(n0 + ty + 8 * u) * CDIM + c0 + tx] = tile[tx][ty + 8 * u];
}

// ---------------------------------------------------------------------------
// Stage 2: QKV projection.  Out(768x1024) = W(768x256) * X|Y(256x1024).
// 4 waves, 64x64 tile, K chunks of 32, double-buffered async LDS staging.
// Q,K written transposed ([n][c]); V natural ([c][n]).
// ---------------------------------------------------------------------------
__global__ __launch_bounds__(128) void k_gemm_qkv(
    const _Float16* __restrict__ Wh, const _Float16* __restrict__ Xt,
    const _Float16* __restrict__ Yt, _Float16* __restrict__ Qt,
    _Float16* __restrict__ Kt, _Float16* __restrict__ Vh) {
    const int nt = blockIdx.x, mt = blockIdx.y, bb = blockIdx.z;
    const int tid = threadIdx.x;
    const int lane = tid & 31, wave = tid >> 5;
    const int g = lane >> 4, ln = lane & 15;

    __shared__ __align__(16) _Float16 As[2][64 * 32];
    __shared__ __align__(16) _Float16 Bs[2][64 * 32];

    const int m0 = mt * 64, n0 = nt * 64;
    const _Float16* srcT = (m0 < 256) ? Xt : Yt;
    const u32x4* w4 = (const u32x4*)Wh;                         // row = 32 u32x4
    const u32x4* s4 = (const u32x4*)srcT + (size_t)bb * NTOK * (CDIM / 8);

    auto stage = [&](int kb, int buf) {
#pragma unroll
        for (int e = 0; e < 2; ++e) {              // 256 u32x4 per tile
            int q = tid + 128 * e;
            int row = q >> 2, seg = q & 3;
            cp16(&As[buf][(row * 4 + seg) * 8], &w4[(m0 + row) * 32 + kb * 4 + seg]);
            cp16(&Bs[buf][(row * 4 + seg) * 8],
                 &s4[(size_t)(n0 + row) * 32 + kb * 4 + seg]);
        }
    };

    v8f acc[4] = {};

    stage(0, 0);
    cp16_wait();
    __syncthreads();

    for (int kb = 0; kb < 8; ++kb) {
        const int cur = kb & 1;
        if (kb + 1 < 8) stage(kb + 1, cur ^ 1);

        v16h a = afrag32(&As[cur][0], wave * 16 + ln, g);
#pragma unroll
        for (int t = 0; t < 4; ++t) {
            v16h bf = bfragT(&Bs[cur][0], 32, t * 16 + ln, g, 0);
            acc[t] = wmma_f16(a, bf, acc[t]);
        }
        if (kb + 1 < 8) cp16_wait();
        __syncthreads();
    }

    // C layout: VGPR r -> row m=r+8g, column n=ln
#pragma unroll
    for (int t = 0; t < 4; ++t) {
#pragma unroll
        for (int r = 0; r < 8; ++r) {
            int row = m0 + wave * 16 + r + 8 * g;
            int col = n0 + t * 16 + ln;
            _Float16 hv = (_Float16)acc[t][r];
            if (row < 256)
                Qt[(size_t)bb * NTOK * CDIM + (size_t)col * CDIM + row] = hv;
            else if (row < 512)
                Kt[(size_t)bb * NTOK * CDIM + (size_t)col * CDIM + (row - 256)] = hv;
            else
                Vh[(size_t)bb * CDIM * NTOK + (size_t)(row - 512) * NTOK + col] = hv;
        }
    }
}

// ---------------------------------------------------------------------------
// Stage 3: l2-normalize over n (reference axis=-1) per (b, h*64+d) column of
// transposed Q/K, folding the x10 scale into Q.  One thread per column.
// ---------------------------------------------------------------------------
__global__ __launch_bounds__(128) void k_l2norm(_Float16* __restrict__ Qt,
                                                _Float16* __restrict__ Kt) {
    const int bb = blockIdx.x >> 2, qq = blockIdx.x & 3;
    const int o = qq * 128 + threadIdx.x;  // 0..511
    _Float16* ptr;
    float mult;
    if (o < 256) { ptr = Qt + (size_t)bb * NTOK * CDIM + o; mult = SCALE; }
    else         { ptr = Kt + (size_t)bb * NTOK * CDIM + (o - 256); mult = 1.0f; }

    float s = 0.0f;
#pragma unroll 8
    for (int n = 0; n < NTOK; ++n) {
        float v = (float)ptr[(size_t)n * CDIM];
        s += v * v;
    }
    float scl = mult / fmaxf(sqrtf(s), 1e-12f);
#pragma unroll 8
    for (int n = 0; n < NTOK; ++n)
        ptr[(size_t)n * CDIM] = (_Float16)((float)ptr[(size_t)n * CDIM] * scl);
}

// ---------------------------------------------------------------------------
// Stage 4: flash attention per (b,h).  4 waves; wave w owns i-rows
// [blockIdx.y*64 + 16w, +16).  Double-buffered 32-wide j blocks:
//   S(16x32) = Q_i^T K_j ; online softmax ; O(16x64) += P * V^T
// One barrier per j-block (P round-trip is intra-wave, DScnt-ordered).
// ---------------------------------------------------------------------------
__global__ __launch_bounds__(128) void k_attention(
    const _Float16* __restrict__ Qt, const _Float16* __restrict__ Kt,
    const _Float16* __restrict__ Vh, _Float16* __restrict__ AOt) {
    const int bb = blockIdx.x >> 2, hh = blockIdx.x & 3;
    const int tid = threadIdx.x;
    const int lane = tid & 31, wave = tid >> 5;
    const int g = lane >> 4, ln = lane & 15;
    const int ibase = blockIdx.y * 64 + wave * 16;

    __shared__ __align__(16) _Float16 KsT[2][32 * 64];  // [j][d] K(=d)-major
    __shared__ __align__(16) _Float16 Vs[2][64 * 32];   // [d][j] K(=j)-major
    __shared__ __align__(16) _Float16 Ps[4 * 16 * 32];

    // Q^T A-fragments (A[m=i][k=d]) straight from global, 4 x b128 pairs
    const _Float16* qrow =
        Qt + (size_t)bb * NTOK * CDIM + (size_t)(ibase + ln) * CDIM + hh * DHEAD;
    v16h aq0 = cat8(*(const v8h*)(qrow + 8 * g),
                    *(const v8h*)(qrow + 16 + 8 * g));
    v16h aq1 = cat8(*(const v8h*)(qrow + 32 + 8 * g),
                    *(const v8h*)(qrow + 48 + 8 * g));

    v8f acc[4] = {};
    float rowm[8], rowl[8];
#pragma unroll
    for (int r = 0; r < 8; ++r) { rowm[r] = -1e30f; rowl[r] = 0.0f; }

    const u32x4* kt4 = (const u32x4*)Kt + (size_t)bb * NTOK * (CDIM / 8);
    const u32x4* vh4 =
        (const u32x4*)Vh + ((size_t)bb * CDIM + hh * DHEAD) * (NTOK / 8);

    auto stageJ = [&](int jb, int buf) {
        const int j0 = jb * 32;
#pragma unroll
        for (int e = 0; e < 2; ++e) {
            int q = tid + 128 * e;
            int rk = q >> 3, sk = q & 7;  // KsT: 8 u32x4 per 64-f16 row
            cp16(&KsT[buf][(rk * 8 + sk) * 8],
                 &kt4[(size_t)(j0 + rk) * 32 + hh * 8 + sk]);
            int rv = q >> 2, sv = q & 3;  // Vs: 4 u32x4 per 32-f16 row
            cp16(&Vs[buf][(rv * 4 + sv) * 8],
                 &vh4[(size_t)rv * 128 + jb * 4 + sv]);
        }
    };

    stageJ(0, 0);
    cp16_wait();
    __syncthreads();

    for (int jb = 0; jb < 32; ++jb) {
        const int cur = jb & 1;
        if (jb + 1 < 32) stageJ(jb + 1, cur ^ 1);

        // S = Q^T K: half t covers j cols [16t,16t+16), two d-chunks of 32
        const _Float16* Kc = &KsT[cur][0];
        v8f s0 = {}, s1 = {};
        s0 = wmma_f16(aq0, bfragT(Kc, 64, ln, g, 0),       s0);
        s0 = wmma_f16(aq1, bfragT(Kc, 64, ln, g, 32),      s0);
        s1 = wmma_f16(aq0, bfragT(Kc, 64, 16 + ln, g, 0),  s1);
        s1 = wmma_f16(aq1, bfragT(Kc, 64, 16 + ln, g, 32), s1);

        // online softmax per C-row (row m=r+8g lives across 16 lanes)
#pragma unroll
        for (int r = 0; r < 8; ++r) {
            float mx = fmaxf(s0[r], s1[r]);
            mx = fmaxf(mx, __shfl_xor(mx, 1));
            mx = fmaxf(mx, __shfl_xor(mx, 2));
            mx = fmaxf(mx, __shfl_xor(mx, 4));
            mx = fmaxf(mx, __shfl_xor(mx, 8));
            float mnew = fmaxf(rowm[r], mx);
            float alpha = __expf(rowm[r] - mnew);
            float p0 = __expf(s0[r] - mnew);
            float p1 = __expf(s1[r] - mnew);
            float ps = p0 + p1;
            ps += __shfl_xor(ps, 1);
            ps += __shfl_xor(ps, 2);
            ps += __shfl_xor(ps, 4);
            ps += __shfl_xor(ps, 8);
            rowl[r] = rowl[r] * alpha + ps;
            rowm[r] = mnew;
            s0[r] = p0;
            s1[r] = p1;
#pragma unroll
            for (int t = 0; t < 4; ++t) acc[t][r] *= alpha;
        }

        // C-layout P -> LDS (per-wave region, intra-wave ordering) -> A-frag
        _Float16* Pw = Ps + wave * 16 * 32;
#pragma unroll
        for (int r = 0; r < 8; ++r) {
            int m = r + 8 * g;
            Pw[m * 32 + ln]      = (_Float16)s0[r];
            Pw[m * 32 + 16 + ln] = (_Float16)s1[r];
        }
        v16h pa = afrag32(Pw, ln, g);

        // O += P * V^T : B[k=jj][n=dd] = Vs[dd][jj] (jj-contiguous per lane)
        const _Float16* Vc = &Vs[cur][0];
#pragma unroll
        for (int t = 0; t < 4; ++t) {
            v16h bv = bfragT(Vc, 32, t * 16 + ln, g, 0);
            acc[t] = wmma_f16(pa, bv, acc[t]);
        }

        if (jb + 1 < 32) cp16_wait();
        __syncthreads();
    }

    // epilogue: O /= l, write AOt[b][i][h*64+d] (K-major for output GEMM)
#pragma unroll
    for (int r = 0; r < 8; ++r) {
        float inv = 1.0f / rowl[r];
        int irow = ibase + r + 8 * g;
        _Float16* dst =
            AOt + (size_t)bb * NTOK * CDIM + (size_t)irow * CDIM + hh * DHEAD;
#pragma unroll
        for (int t = 0; t < 4; ++t)
            dst[t * 16 + ln] = (_Float16)(acc[t][r] * inv);
    }
}

// ---------------------------------------------------------------------------
// Stage 5: output projection.  out(256x1024) = Wo(256x256)*AO(256x1024)+bias.
// ---------------------------------------------------------------------------
__global__ __launch_bounds__(128) void k_gemm_out(
    const _Float16* __restrict__ Wh, const _Float16* __restrict__ AOt,
    const float* __restrict__ bias, float* __restrict__ out) {
    const int nt = blockIdx.x, mt = blockIdx.y, bb = blockIdx.z;
    const int tid = threadIdx.x;
    const int lane = tid & 31, wave = tid >> 5;
    const int g = lane >> 4, ln = lane & 15;

    __shared__ __align__(16) _Float16 As[2][64 * 32];
    __shared__ __align__(16) _Float16 Bs[2][64 * 32];

    const int m0 = mt * 64, n0 = nt * 64;
    const u32x4* w4 = (const u32x4*)Wh;
    const u32x4* s4 = (const u32x4*)AOt + (size_t)bb * NTOK * (CDIM / 8);

    auto stage = [&](int kb, int buf) {
#pragma unroll
        for (int e = 0; e < 2; ++e) {
            int q = tid + 128 * e;
            int row = q >> 2, seg = q & 3;
            cp16(&As[buf][(row * 4 + seg) * 8], &w4[(m0 + row) * 32 + kb * 4 + seg]);
            cp16(&Bs[buf][(row * 4 + seg) * 8],
                 &s4[(size_t)(n0 + row) * 32 + kb * 4 + seg]);
        }
    };

    v8f acc[4] = {};

    stage(0, 0);
    cp16_wait();
    __syncthreads();

    for (int kb = 0; kb < 8; ++kb) {
        const int cur = kb & 1;
        if (kb + 1 < 8) stage(kb + 1, cur ^ 1);

        v16h a = afrag32(&As[cur][0], wave * 16 + ln, g);
#pragma unroll
        for (int t = 0; t < 4; ++t) {
            v16h bf = bfragT(&Bs[cur][0], 32, t * 16 + ln, g, 0);
            acc[t] = wmma_f16(a, bf, acc[t]);
        }
        if (kb + 1 < 8) cp16_wait();
        __syncthreads();
    }

#pragma unroll
    for (int t = 0; t < 4; ++t) {
#pragma unroll
        for (int r = 0; r < 8; ++r) {
            int row = m0 + wave * 16 + r + 8 * g;
            int col = n0 + t * 16 + ln;
            out[(size_t)bb * CDIM * NTOK + (size_t)row * NTOK + col] =
                acc[t][r] + bias[row];
        }
    }
}

// ---------------------------------------------------------------------------
// Launch
// ---------------------------------------------------------------------------
extern "C" void kernel_launch(void* const* d_in, const int* in_sizes, int n_in,
                              void* d_out, int out_size, void* d_ws,
                              size_t ws_size, hipStream_t stream) {
    const float* x     = (const float*)d_in[0];
    const float* y     = (const float*)d_in[1];
    const float* w_qkv = (const float*)d_in[2];
    const float* w_out = (const float*)d_in[3];
    const float* b_out = (const float*)d_in[4];
    float* out = (float*)d_out;

    char* ws = (char*)d_ws;
    size_t off = 0;
    const size_t ACT = (size_t)BATCH * CDIM * NTOK * 2;  // 8 MiB each
    _Float16* Wh  = (_Float16*)(ws + off); off += (size_t)768 * 256 * 2;
    _Float16* WOh = (_Float16*)(ws + off); off += (size_t)256 * 256 * 2;
    _Float16* Xt  = (_Float16*)(ws + off); off += ACT;
    _Float16* Yt  = (_Float16*)(ws + off); off += ACT;
    _Float16* Qt  = (_Float16*)(ws + off); off += ACT;
    _Float16* Kt  = (_Float16*)(ws + off); off += ACT;
    _Float16* Vh  = (_Float16*)(ws + off); off += ACT;
    _Float16* AOt = (_Float16*)(ws + off); off += ACT;

    k_f32_to_f16<<<768, 256, 0, stream>>>(w_qkv, Wh, 768 * 256);
    k_f32_to_f16<<<256, 256, 0, stream>>>(w_out, WOh, 256 * 256);
    k_transpose_f16<<<dim3(32, 8, 16), dim3(32, 8), 0, stream>>>(x, Xt);
    k_transpose_f16<<<dim3(32, 8, 16), dim3(32, 8), 0, stream>>>(y, Yt);

    k_gemm_qkv<<<dim3(16, 12, 16), 128, 0, stream>>>(Wh, Xt, Yt, Qt, Kt, Vh);
    k_l2norm<<<BATCH * 4, 128, 0, stream>>>(Qt, Kt);
    k_attention<<<dim3(BATCH * HEADS, 16), 128, 0, stream>>>(Qt, Kt, Vh, AOt);
    k_gemm_out<<<dim3(16, 4, 16), 128, 0, stream>>>(WOh, AOt, b_out, out);
}
